// Detic_26147760898062
// MI455X (gfx1250) — compile-verified, matrix operand order checked by reading, not run on hardware
//
#include <hip/hip_runtime.h>
#include <hip/hip_bf16.h>
#include <math.h>

// ---------------- constants ----------------
#define NPROP   512
#define DFEAT   12544      // 7*7*256
#define H1      1024
#define H2      1024
#define CLSD    512
#define NCLS    1204       // 1203 fg + 1 bg
#define NFG     1203
#define IMGSZ   896.0f
#define SCALE_CLAMP 4.135166556742356f   // log(1000/16)
#define TOPK    300

typedef __attribute__((ext_vector_type(16))) __bf16        v16bf;
typedef __attribute__((ext_vector_type(8)))  float         v8f;
typedef __attribute__((ext_vector_type(4)))  unsigned int  u32x4;
typedef __attribute__((ext_vector_type(8)))  int           i32x8;
typedef __attribute__((ext_vector_type(4)))  int           i32x4;

// Tensor Data Mover availability (probe: builtin exists on both toolchains,
// arity differs: clang-22/ROCm7.2 = 5 args, clang-23/therock = 6 args)
#if defined(__has_builtin)
#  if __has_builtin(__builtin_amdgcn_tensor_load_to_lds)
#    define HAVE_TDM 1
#  endif
#endif
#ifndef HAVE_TDM
#  define HAVE_TDM 0
#endif

// ================= WMMA GEMM =================
// C[M,N] = epilogue(A[M,K] * B[K,N]); fp32 in memory, bf16 matrix pipe, fp32 acc.
// Block: 256 thr = 8 waves; block tile 64(M) x 128(N); K step 32.
// Wave tile 32x32 = 2x2 WMMA fragments (each fragment reused twice).
// A tile staged by TDM (tensor_load_to_lds, 64x32 f32, row stride K);
// B tile staged cooperatively with bf16 transpose (WMMA B wants K-contig/lane).
// EPI: 0 = +bias, 1 = +bias,relu, 2 = C += sigmoid(acc)
#define TM 64
#define TN 128
#define TK 32

template<int EPI>
__global__ __launch_bounds__(256)
void gemm_wmma(const float* __restrict__ A, const float* __restrict__ B,
               const float* __restrict__ bias, float* __restrict__ C,
               int M, int N, int K) {
  __shared__ float  lAf[TM][TK];          // 8 KB, filled by TDM (row-major pack)
  __shared__ __bf16 lBt[TN][TK + 16];     // 12 KB, 96B rows -> 16B-aligned groups

  const int tid  = threadIdx.x;
  const int m0   = blockIdx.y * TM;
  const int n0   = blockIdx.x * TN;
  const int wave = tid >> 5;
  const int lane = tid & 31;
  const int half = lane >> 4;           // K-group select per ISA 16-bit layout
  const int l15  = lane & 15;
  const int wm   = wave >> 2;           // 0..1 -> M offset wm*32
  const int wn   = wave & 3;            // 0..3 -> N offset wn*32

  v8f acc[2][2] = {{{}, {}}, {{}, {}}};

#if HAVE_TDM
  const unsigned ldsA = (unsigned)(uintptr_t)(&lAf[0][0]);
#endif

  for (int kk = 0; kk < K; kk += TK) {
    __syncthreads();

#if HAVE_TDM
    // ---- A tile via Tensor Data Mover: 2D tile 32(dim0) x 64(dim1) f32 ----
    if (tid < 32) {
      unsigned long long ga =
          (unsigned long long)(uintptr_t)(A + (size_t)m0 * K + kk);
      // D# group0: count=1 | lds_addr | global_addr[56:0] | type=2
      u32x4 g0 = { 1u, ldsA, (unsigned)(ga & 0xffffffffu),
                   (unsigned)((ga >> 32) & 0x01ffffffu) | 0x80000000u };
      // D# group1: data_size=4B; tensor_dim0=K; tensor_dim1=M;
      //            tile_dim0=TK; tile_dim1=TM; dim0_stride=K
      i32x8 g1 = { (int)(2u << 16),
                   (int)(((unsigned)K & 0xffffu) << 16),
                   (int)((((unsigned)K >> 16) & 0xffffu) |
                         (((unsigned)M & 0xffffu) << 16)),
                   (int)((((unsigned)M >> 16) & 0xffffu) | ((unsigned)TK << 16)),
                   (int)TM,
                   (int)K,
                   0, 0 };
      i32x4 g2 = {}, g3 = {};
#if __clang_major__ >= 23
      i32x8 g4 = {};
      __builtin_amdgcn_tensor_load_to_lds(g0, g1, g2, g3, g4, 0);
#else
      __builtin_amdgcn_tensor_load_to_lds(g0, g1, g2, g3, 0);
#endif
    }
#else
    // ---- fallback: cooperative A staging, 2 float4 per thread ----
    #pragma unroll
    for (int t = 0; t < 2; ++t) {
      int f = (tid << 3) + (t << 2);   // 2048 floats
      int r = f >> 5, c = f & 31;
      *reinterpret_cast<float4*>(&lAf[r][c]) =
          *reinterpret_cast<const float4*>(A + (size_t)(m0 + r) * K + kk + c);
    }
#endif

    // ---- B tile 32x128 f32 -> transpose into bf16 LDS, 4 float4/thread ----
    #pragma unroll
    for (int t = 0; t < 4; ++t) {
      int f = (tid << 4) + (t << 2);   // 4096 floats
      int r = f >> 7, c = f & 127;
      float4 v = make_float4(0.f, 0.f, 0.f, 0.f);
      if (n0 + c < N) {
        v = *reinterpret_cast<const float4*>(B + (size_t)(kk + r) * N + n0 + c);
        if (kk + TK < K)   // prefetch next K-tile of the weight stream
          __builtin_prefetch(B + (size_t)(kk + TK + r) * N + n0 + c, 0, 1);
      }
      lBt[c+0][r] = (__bf16)v.x; lBt[c+1][r] = (__bf16)v.y;
      lBt[c+2][r] = (__bf16)v.z; lBt[c+3][r] = (__bf16)v.w;
    }

#if HAVE_TDM
    if (tid < 32) __builtin_amdgcn_s_wait_tensorcnt(0);
#endif
    __syncthreads();

    // ---- fragments per ISA 16-bit A/B layouts ----
    // lanes 0-15: K = e (e<8), 16+(e-8) (e>=8); lanes 16-31: +8
    const int kb = half << 3;
    v16bf a[2], b[2];
    #pragma unroll
    for (int mi = 0; mi < 2; ++mi) {
      const float* row = &lAf[wm * 32 + mi * 16 + l15][0];
      #pragma unroll
      for (int e = 0; e < 8; ++e) {
        a[mi][e]     = (__bf16)row[kb + e];
        a[mi][e + 8] = (__bf16)row[kb + 16 + e];
      }
    }
    #pragma unroll
    for (int ni = 0; ni < 2; ++ni) {
      const __bf16* col = &lBt[wn * 32 + ni * 16 + l15][0];
      #pragma unroll
      for (int e = 0; e < 8; ++e) {
        b[ni][e]     = col[kb + e];
        b[ni][e + 8] = col[kb + 16 + e];
      }
    }
    #pragma unroll
    for (int mi = 0; mi < 2; ++mi)
      #pragma unroll
      for (int ni = 0; ni < 2; ++ni)
        acc[mi][ni] = __builtin_amdgcn_wmma_f32_16x16x32_bf16(
            false, a[mi], false, b[ni], (short)0, acc[mi][ni], false, false);
  }

  // ---- epilogue: C/D layout: VGPR r -> M = +r+half*8, N = l15 ----
  #pragma unroll
  for (int mi = 0; mi < 2; ++mi) {
    #pragma unroll
    for (int ni = 0; ni < 2; ++ni) {
      #pragma unroll
      for (int r = 0; r < 8; ++r) {
        int m = m0 + wm * 32 + mi * 16 + r + (half << 3);
        int n = n0 + wn * 32 + ni * 16 + l15;
        if (n < N) {
          float v = acc[mi][ni][r];
          if (EPI == 0) {
            C[(size_t)m * N + n] = v + bias[n];
          } else if (EPI == 1) {
            C[(size_t)m * N + n] = fmaxf(v + bias[n], 0.0f);
          } else {
            C[(size_t)m * N + n] += 1.0f / (1.0f + __expf(-v));
          }
        }
      }
    }
  }
}

// ================= ROI align =================
__global__ __launch_bounds__(256)
void roi_align_kernel(const float* __restrict__ p2, const float* __restrict__ p3,
                      const float* __restrict__ p4, const float* __restrict__ p5,
                      const float* __restrict__ boxes, float* __restrict__ xout) {
  const int p = blockIdx.x;
  const float bx1 = boxes[p*4+0], by1 = boxes[p*4+1];
  const float bx2 = boxes[p*4+2], by2 = boxes[p*4+3];
  float area = fmaxf((bx2 - bx1) * (by2 - by1), 1e-8f);
  float lv = floorf(4.0f + log2f(sqrtf(area) / 224.0f + 1e-8f));
  lv = fminf(fmaxf(lv, 2.0f), 5.0f);
  const int lvl = (int)lv - 2;
  const float* feat; int H, W; float stride;
  if      (lvl == 0) { feat = p2; H = 224; W = 224; stride =  4.f; }
  else if (lvl == 1) { feat = p3; H = 112; W = 112; stride =  8.f; }
  else if (lvl == 2) { feat = p4; H =  56; W =  56; stride = 16.f; }
  else               { feat = p5; H =  28; W =  28; stride = 32.f; }
  const float s  = 1.0f / stride;
  const float x1 = bx1 * s - 0.5f, y1 = by1 * s - 0.5f;
  const float x2 = bx2 * s - 0.5f, y2 = by2 * s - 0.5f;
  const float bw = (x2 - x1) / 7.0f, bh = (y2 - y1) / 7.0f;

  for (int o = threadIdx.x; o < DFEAT; o += 256) {
    int c  = o & 255;
    int pp = o >> 8;
    int px = pp % 7, py = pp / 7;
    float acc = 0.f;
    #pragma unroll
    for (int sy = 0; sy < 2; ++sy) {
      float ty = ((float)(py * 2 + sy) + 0.5f) * 0.5f;
      float yy = fminf(fmaxf(y1 + ty * bh, 0.f), (float)(H - 1));
      int   yi0 = (int)floorf(yy);
      int   yi1 = min(yi0 + 1, H - 1);
      float wy  = yy - floorf(yy);
      #pragma unroll
      for (int sx = 0; sx < 2; ++sx) {
        float tx = ((float)(px * 2 + sx) + 0.5f) * 0.5f;
        float xx = fminf(fmaxf(x1 + tx * bw, 0.f), (float)(W - 1));
        int   xi0 = (int)floorf(xx);
        int   xi1 = min(xi0 + 1, W - 1);
        float wx  = xx - floorf(xx);
        float f00 = feat[((size_t)yi0 * W + xi0) * 256 + c];
        float f01 = feat[((size_t)yi0 * W + xi1) * 256 + c];
        float f10 = feat[((size_t)yi1 * W + xi0) * 256 + c];
        float f11 = feat[((size_t)yi1 * W + xi1) * 256 + c];
        acc += f00 * (1.f - wy) * (1.f - wx) + f01 * (1.f - wy) * wx
             + f10 * wy * (1.f - wx)         + f11 * wy * wx;
      }
    }
    xout[(size_t)p * DFEAT + o] = acc * 0.25f;
  }
}

// ================= small kernels =================
__global__ void zero_kernel(float* __restrict__ p, int n) {
  int i = blockIdx.x * blockDim.x + threadIdx.x;
  if (i < n) p[i] = 0.0f;
}

__global__ void copy_boxes_kernel(const float* __restrict__ src, float* __restrict__ dst) {
  int i = blockIdx.x * blockDim.x + threadIdx.x;
  if (i < NPROP * 4) dst[i] = src[i];
}

// column-normalize zs_weight (norm over 512 feature rows)
__global__ void zsnorm_kernel(const float* __restrict__ zs, float* __restrict__ zsn) {
  int c = blockIdx.x * blockDim.x + threadIdx.x;
  if (c >= NCLS) return;
  float s = 0.f;
  for (int r = 0; r < CLSD; ++r) { float v = zs[(size_t)r * NCLS + c]; s += v * v; }
  float inv = 1.0f / fmaxf(sqrtf(s), 1e-12f);
  for (int r = 0; r < CLSD; ++r)
    zsn[(size_t)r * NCLS + c] = zs[(size_t)r * NCLS + c] * inv;
}

// xn = TEMP * xf / max(||xf||_row, eps)
__global__ __launch_bounds__(256)
void rownorm_kernel(const float* __restrict__ xf, float* __restrict__ xn) {
  __shared__ float red[256];
  int p = blockIdx.x;
  float s = 0.f;
  for (int i = threadIdx.x; i < CLSD; i += 256) { float v = xf[(size_t)p*CLSD+i]; s += v*v; }
  red[threadIdx.x] = s; __syncthreads();
  for (int st = 128; st > 0; st >>= 1) {
    if (threadIdx.x < st) red[threadIdx.x] += red[threadIdx.x + st];
    __syncthreads();
  }
  float sc = 50.0f / fmaxf(sqrtf(red[0]), 1e-12f);
  for (int i = threadIdx.x; i < CLSD; i += 256)
    xn[(size_t)p*CLSD+i] = xf[(size_t)p*CLSD+i] * sc;
}

// deltas = h @ bbox2_w + bbox2_b, then apply + clip; in-place on boxes
__global__ __launch_bounds__(256)
void deltas_kernel(const float* __restrict__ h, const float* __restrict__ W2,
                   const float* __restrict__ b2, float* __restrict__ boxes,
                   float wx, float wy, float ww, float wh) {
  __shared__ float red[256];
  __shared__ float d[4];
  int p = blockIdx.x;
  #pragma unroll
  for (int j = 0; j < 4; ++j) {
    float s = 0.f;
    for (int k = threadIdx.x; k < H1; k += 256)
      s += h[(size_t)p * H1 + k] * W2[(size_t)k * 4 + j];
    red[threadIdx.x] = s; __syncthreads();
    for (int st = 128; st > 0; st >>= 1) {
      if (threadIdx.x < st) red[threadIdx.x] += red[threadIdx.x + st];
      __syncthreads();
    }
    if (threadIdx.x == 0) d[j] = red[0] + b2[j];
    __syncthreads();
  }
  if (threadIdx.x == 0) {
    float x1 = boxes[p*4+0], y1 = boxes[p*4+1], x2 = boxes[p*4+2], y2 = boxes[p*4+3];
    float wdt = x2 - x1, hgt = y2 - y1;
    float cx = x1 + 0.5f * wdt, cy = y1 + 0.5f * hgt;
    float dx = d[0] / wx, dy = d[1] / wy;
    float dw = fminf(d[2] / ww, SCALE_CLAMP), dh = fminf(d[3] / wh, SCALE_CLAMP);
    float pcx = dx * wdt + cx, pcy = dy * hgt + cy;
    float pw = __expf(dw) * wdt, ph = __expf(dh) * hgt;
    boxes[p*4+0] = fminf(fmaxf(pcx - 0.5f * pw, 0.f), IMGSZ);
    boxes[p*4+1] = fminf(fmaxf(pcy - 0.5f * ph, 0.f), IMGSZ);
    boxes[p*4+2] = fminf(fmaxf(pcx + 0.5f * pw, 0.f), IMGSZ);
    boxes[p*4+3] = fminf(fmaxf(pcy + 0.5f * ph, 0.f), IMGSZ);
  }
}

// per-proposal: scores = sqrt(mean_prob * prop_score); fg max/argmax; cand
__global__ __launch_bounds__(256)
void score_kernel(const float* __restrict__ prob_sum, const float* __restrict__ pscores,
                  float* __restrict__ cand, int* __restrict__ clsidx) {
  __shared__ float rv[256];
  __shared__ int   ri[256];
  int p = blockIdx.x;
  float ps = pscores[p];
  float bestv = -1.f; int besti = 0;
  for (int c = threadIdx.x; c < NFG; c += 256) {
    float sc = sqrtf(prob_sum[(size_t)p * NCLS + c] * (1.0f / 3.0f) * ps);
    if (sc > bestv) { bestv = sc; besti = c; }
  }
  rv[threadIdx.x] = bestv; ri[threadIdx.x] = besti; __syncthreads();
  for (int st = 128; st > 0; st >>= 1) {
    if (threadIdx.x < st) {
      float ov = rv[threadIdx.x + st]; int oi = ri[threadIdx.x + st];
      if (ov > rv[threadIdx.x] || (ov == rv[threadIdx.x] && oi < ri[threadIdx.x])) {
        rv[threadIdx.x] = ov; ri[threadIdx.x] = oi;
      }
    }
    __syncthreads();
  }
  if (threadIdx.x == 0) {
    clsidx[p] = ri[0];
    cand[p]   = rv[0] > 0.5f ? rv[0] : -1.0f;
  }
}

// bitonic argsort (descending, ties -> lower index) of 512 candidates
__global__ __launch_bounds__(512)
void sort_kernel(const float* __restrict__ cand, int* __restrict__ order,
                 float* __restrict__ skey) {
  __shared__ float key[NPROP];
  __shared__ int   idx[NPROP];
  int t = threadIdx.x;
  key[t] = cand[t]; idx[t] = t;
  __syncthreads();
  for (int ksz = 2; ksz <= NPROP; ksz <<= 1) {
    for (int j = ksz >> 1; j > 0; j >>= 1) {
      int ixj = t ^ j;
      if (ixj > t) {
        float a = key[t], b = key[ixj];
        int   ia = idx[t], ib = idx[ixj];
        bool gt = (a < b) || (a == b && ia > ib); // descending / quasi-stable
        bool up = ((t & ksz) == 0);
        if (up ? gt : !gt) {
          key[t] = b; key[ixj] = a;
          idx[t] = ib; idx[ixj] = ia;
        }
      }
      __syncthreads();
    }
  }
  order[t] = idx[t]; skey[t] = key[t];
}

// greedy NMS with per-class box offsets (matches reference fori_loop)
__global__ __launch_bounds__(512)
void nms_kernel(const float* __restrict__ boxes, const int* __restrict__ clsidx,
                const int* __restrict__ order, const float* __restrict__ skey,
                float* __restrict__ final_sc) {
  __shared__ float bx[NPROP][4];
  __shared__ int keep[NPROP];
  int t = threadIdx.x;
  int o = order[t];
  float off = (float)clsidx[o] * (IMGSZ + 1.0f);
  bx[t][0] = boxes[o*4+0] + off; bx[t][1] = boxes[o*4+1] + off;
  bx[t][2] = boxes[o*4+2] + off; bx[t][3] = boxes[o*4+3] + off;
  keep[t] = 1;
  __syncthreads();
  for (int i = 0; i < NPROP; ++i) {
    if (keep[i] && t > i && keep[t]) {
      float ax1 = bx[i][0], ay1 = bx[i][1], ax2 = bx[i][2], ay2 = bx[i][3];
      float cx1 = bx[t][0], cy1 = bx[t][1], cx2 = bx[t][2], cy2 = bx[t][3];
      float aarea = (ax2 - ax1) * (ay2 - ay1);
      float carea = (cx2 - cx1) * (cy2 - cy1);
      float iw = fmaxf(fminf(ax2, cx2) - fmaxf(ax1, cx1), 0.f);
      float ih = fmaxf(fminf(ay2, cy2) - fmaxf(ay1, cy1), 0.f);
      float inter = iw * ih;
      float iou = inter / (aarea + carea - inter + 1e-9f);
      if (iou > 0.5f) keep[t] = 0;
    }
    __syncthreads();
  }
  final_sc[t] = (keep[t] && skey[t] > 0.f) ? skey[t] : -1.0f;
}

// top-300 of `final` (kept entries are already in descending order)
__global__ void select_kernel(const float* __restrict__ final_sc,
                              int* __restrict__ tsi, float* __restrict__ tsc) {
  if (threadIdx.x != 0 || blockIdx.x != 0) return;
  int cnt = 0;
  for (int j = 0; j < NPROP && cnt < TOPK; ++j)
    if (final_sc[j] > -1.0f) { tsi[cnt] = j; tsc[cnt] = final_sc[j]; ++cnt; }
  for (int j = 0; j < NPROP && cnt < TOPK; ++j)
    if (!(final_sc[j] > -1.0f)) { tsi[cnt] = j; tsc[cnt] = -1.0f; ++cnt; }
}

// gather outputs: [boxes(300,4) | scores(300) | cls(300) | xf(300,512) | pscores(300)]
__global__ __launch_bounds__(256)
void gather_kernel(const float* __restrict__ boxes, const float* __restrict__ pscores,
                   const int* __restrict__ clsidx, const float* __restrict__ xf,
                   const int* __restrict__ tsi, const int* __restrict__ order,
                   const float* __restrict__ tsc, float* __restrict__ out) {
  int r = blockIdx.x;            // 0..299
  int orig = order[tsi[r]];
  if (threadIdx.x == 0) {
    out[r*4+0] = boxes[orig*4+0]; out[r*4+1] = boxes[orig*4+1];
    out[r*4+2] = boxes[orig*4+2]; out[r*4+3] = boxes[orig*4+3];
    out[1200 + r]   = tsc[r];
    out[1500 + r]   = (float)clsidx[orig];
    out[155400 + r] = pscores[orig];
  }
  for (int i = threadIdx.x; i < CLSD; i += 256)
    out[1800 + (size_t)r * CLSD + i] = xf[(size_t)orig * CLSD + i];
}

// ================= host =================
extern "C" void kernel_launch(void* const* d_in, const int* in_sizes, int n_in,
                              void* d_out, int out_size, void* d_ws, size_t ws_size,
                              hipStream_t stream) {
  const float* p2   = (const float*)d_in[0];
  const float* p3   = (const float*)d_in[1];
  const float* p4   = (const float*)d_in[2];
  const float* p5   = (const float*)d_in[3];
  const float* pbox = (const float*)d_in[4];
  const float* psc  = (const float*)d_in[5];
  const float* fc1w = (const float*)d_in[6];
  const float* fc1b = (const float*)d_in[7];
  const float* fc2w = (const float*)d_in[8];
  const float* fc2b = (const float*)d_in[9];
  const float* clsw = (const float*)d_in[10];
  const float* clsb = (const float*)d_in[11];
  const float* zsw  = (const float*)d_in[12];
  const float* bb1w = (const float*)d_in[13];
  const float* bb1b = (const float*)d_in[14];
  const float* bb2w = (const float*)d_in[15];
  const float* bb2b = (const float*)d_in[16];
  float* out = (float*)d_out;

  // workspace carving (256B aligned)
  char* ws = (char*)d_ws;
  size_t off = 0;
  auto carve = [&](size_t bytes) -> char* {
    char* p = ws + off;
    off = (off + bytes + 255) & ~(size_t)255;
    return p;
  };
  float* x0       = (float*)carve((size_t)NPROP * DFEAT * 4);
  float* x1       = (float*)carve((size_t)NPROP * H1 * 4);
  float* x2       = (float*)carve((size_t)NPROP * H2 * 4);
  float* xf       = (float*)carve((size_t)NPROP * CLSD * 4);
  float* xn       = (float*)carve((size_t)NPROP * CLSD * 4);
  float* hbb      = (float*)carve((size_t)NPROP * H1 * 4);
  float* zsn      = (float*)carve((size_t)CLSD * NCLS * 4);
  float* prob_sum = (float*)carve((size_t)NPROP * NCLS * 4);
  float* cboxes   = (float*)carve((size_t)NPROP * 4 * 4);
  float* cand     = (float*)carve((size_t)NPROP * 4);
  int*   clsidx   = (int*)  carve((size_t)NPROP * 4);
  int*   order    = (int*)  carve((size_t)NPROP * 4);
  float* skey     = (float*)carve((size_t)NPROP * 4);
  float* final_sc = (float*)carve((size_t)NPROP * 4);
  int*   tsi      = (int*)  carve((size_t)TOPK * 4);
  float* tsc      = (float*)carve((size_t)TOPK * 4);

  // one-time prep
  zero_kernel<<<(NPROP * NCLS + 255) / 256, 256, 0, stream>>>(prob_sum, NPROP * NCLS);
  zsnorm_kernel<<<(NCLS + 255) / 256, 256, 0, stream>>>(zsw, zsn);
  copy_boxes_kernel<<<(NPROP * 4 + 255) / 256, 256, 0, stream>>>(pbox, cboxes);

  const float dwts[3][4] = {{10.f,10.f,5.f,5.f},{20.f,20.f,10.f,10.f},{30.f,30.f,15.f,15.f}};

  for (int k = 0; k < 3; ++k) {
    // ROI align over FPN levels -> x0 (512 x 12544)
    roi_align_kernel<<<NPROP, 256, 0, stream>>>(p2, p3, p4, p5, cboxes, x0);

    // fc1: (512 x 12544) @ (12544 x 1024), bias+relu  -- dominant GEMM
    gemm_wmma<1><<<dim3(H1 / TN, NPROP / TM), 256, 0, stream>>>(
        x0, fc1w + (size_t)k * DFEAT * H1, fc1b + (size_t)k * H1, x1,
        NPROP, H1, DFEAT);
    // fc2: (512 x 1024) @ (1024 x 1024), bias+relu
    gemm_wmma<1><<<dim3(H2 / TN, NPROP / TM), 256, 0, stream>>>(
        x1, fc2w + (size_t)k * H1 * H2, fc2b + (size_t)k * H2, x2,
        NPROP, H2, H1);
    // cls head: (512 x 1024) @ (1024 x 512), bias
    gemm_wmma<0><<<dim3(CLSD / TN, NPROP / TM), 256, 0, stream>>>(
        x2, clsw + (size_t)k * H2 * CLSD, clsb + (size_t)k * CLSD, xf,
        NPROP, CLSD, H2);
    // xn = 50 * xf / ||xf||
    rownorm_kernel<<<NPROP, 256, 0, stream>>>(xf, xn);
    // prob_sum += sigmoid(xn @ zs_norm): (512 x 512) @ (512 x 1204)
    gemm_wmma<2><<<dim3((NCLS + TN - 1) / TN, NPROP / TM), 256, 0, stream>>>(
        xn, zsn, nullptr, prob_sum, NPROP, NCLS, CLSD);
    // bbox branch
    gemm_wmma<1><<<dim3(H1 / TN, NPROP / TM), 256, 0, stream>>>(
        x2, bb1w + (size_t)k * H2 * H1, bb1b + (size_t)k * H1, hbb,
        NPROP, H1, H2);
    deltas_kernel<<<NPROP, 256, 0, stream>>>(
        hbb, bb2w + (size_t)k * H1 * 4, bb2b + (size_t)k * 4, cboxes,
        dwts[k][0], dwts[k][1], dwts[k][2], dwts[k][3]);
  }

  // scoring, per-class NMS, top-300, gather
  score_kernel <<<NPROP, 256, 0, stream>>>(prob_sum, psc, cand, clsidx);
  sort_kernel  <<<1, NPROP, 0, stream>>>(cand, order, skey);
  nms_kernel   <<<1, NPROP, 0, stream>>>(cboxes, clsidx, order, skey, final_sc);
  select_kernel<<<1, 1, 0, stream>>>(final_sc, tsi, tsc);
  gather_kernel<<<TOPK, 256, 0, stream>>>(cboxes, psc, clsidx, xf, tsi, order, tsc, out);
}
